// MCDropoutLinkPredictor_55490977465143
// MI455X (gfx1250) — matrix operand order repaired and checked by reading, not compiled, not run on hardware
//
#include <hip/hip_runtime.h>
#include <hip/hip_bf16.h>

typedef __attribute__((ext_vector_type(16))) __bf16 v16bf;
typedef __attribute__((ext_vector_type(8)))  float  v8f;
typedef __attribute__((ext_vector_type(4)))  unsigned int v4u;
typedef __attribute__((ext_vector_type(8)))  int v8i;
typedef __attribute__((ext_vector_type(4)))  int v4i;

#define DIN 256
#define DH  128

#if defined(__HIP_DEVICE_COMPILE__) && __has_builtin(__builtin_amdgcn_tensor_load_to_lds)
#define HAVE_TDM 1
#else
#define HAVE_TDM 0
#endif

// ---------------- degree / normalization ----------------

__global__ void zero_f32_kernel(float* __restrict__ p, int n) {
    int i = blockIdx.x * blockDim.x + threadIdx.x;
    if (i < n) p[i] = 0.0f;
}

__global__ void count_deg_kernel(const int* __restrict__ dst, float* __restrict__ deg, int E) {
    int e = blockIdx.x * blockDim.x + threadIdx.x;
    if (e < E) unsafeAtomicAdd(&deg[dst[e]], 1.0f);
}

__global__ void deg_to_dinv_kernel(float* __restrict__ deg, int n) {
    int i = blockIdx.x * blockDim.x + threadIdx.x;
    if (i < n) deg[i] = rsqrtf(deg[i] + 1.0f);   // self-loop included
}

// ---------------- W pre-pack: bf16, B-fragment order ----------------
// pk[((kb*8 + w)*32 + lane)*16 + j] = bf16( W[(kb*32 + (j<8 ? (lane>>4)*8 + j : 16 + (lane>>4)*8 + j-8))*DH + w*16 + (lane&15)] )
// so in the GEMM each lane's 16 B values are one contiguous 32-byte read.

__global__ void pack_w_kernel(const float* __restrict__ W, __bf16* __restrict__ pk, int KDw) {
    int t = blockIdx.x * blockDim.x + threadIdx.x;
    if (t >= KDw * DH) return;
    int j    = t & 15;
    int lane = (t >> 4) & 31;
    int w    = (t >> 9) & 7;
    int kb   = t >> 12;
    int k = kb * 32 + ((j < 8) ? ((lane >> 4) * 8 + j) : (16 + (lane >> 4) * 8 + (j - 8)));
    int c = w * 16 + (lane & 15);
    pk[t] = (__bf16)W[k * DH + c];
}

// ---------------- WMMA bf16 GEMM: out[nrows,128] = op(A)[nrows,KD] @ W[KD,128] ----------------
// Block: 256 threads = 8 waves; block owns 32 rows (two 16-row tiles, dual
// accumulators so back-to-back independent WMMAs hide the WMMA->VALU hazard);
// wave w owns cols [16w,16w+16). The 32 x KD fp32 A tile is staged into LDS by
// the Tensor Data Mover (wave 0 issues tensor_load_to_lds with a 2D D#;
// pad_enable inserts 4 DWORDs per row -> row stride KD+4, no bank conflicts).

template <int KD, bool RELU>
__global__ __launch_bounds__(256) void gemm_wmma_bf16_kernel(
    const float* __restrict__ A, const __bf16* __restrict__ pkW,
    float* __restrict__ out, int nrows)
{
    constexpr int PAD = 4;
    constexpr int LDA = KD + PAD;
    __shared__ float atile[32 * LDA];

    const int lane  = threadIdx.x & 31;
    const int wave  = threadIdx.x >> 5;
    const int m     = lane & 15;
    const int khalf = lane >> 4;
    const int rowBase = blockIdx.x * 32;
    const float* gA = A + (long)rowBase * KD;

#if HAVE_TDM
    if (wave == 0) {
        unsigned long long ga = (unsigned long long)(const void*)gA;
        unsigned lds = (unsigned)(unsigned long long)(const void*)&atile[0]; // low 32b = LDS addr
        int rowsLeft = nrows - rowBase; if (rowsLeft > 32) rowsLeft = 32;
        v4u g0;
        g0.x = 1u;                                            // count=1, user descriptor
        g0.y = lds;                                           // LDS byte address
        g0.z = (unsigned)ga;                                  // global_addr[31:0]
        g0.w = (unsigned)((ga >> 32) & 0x1FFFFFFull) | (2u << 30); // addr[56:32] | type=2
        const unsigned padIvl = (KD == 256) ? 7u : 6u;        // pad after 2^(v+1) dwords = KD
        v8i g1;
        g1[0] = (int)((2u << 16) | (1u << 20) | (padIvl << 22) | (3u << 25)); // 4B elems, pad 4 dw
        g1[1] = (int)(((unsigned)KD & 0xFFFFu) << 16);                        // tensor_dim0 lo16
        g1[2] = (int)(((unsigned)KD >> 16) | (((unsigned)rowsLeft & 0xFFFFu) << 16)); // td0 hi | td1 lo
        g1[3] = (int)(((unsigned)KD) << 16);                                  // td1 hi=0 | tile_dim0
        g1[4] = 32;                                                           // tile_dim1=32, tile_dim2=0
        g1[5] = KD;                                                           // tensor_dim0_stride lo32
        g1[6] = 0;
        g1[7] = 0;
        v4i z4 = {0, 0, 0, 0};
#if __clang_major__ >= 23
        v8i z8 = {0, 0, 0, 0, 0, 0, 0, 0};
        __builtin_amdgcn_tensor_load_to_lds(g0, g1, z4, z4, z8, 0);
#else
        __builtin_amdgcn_tensor_load_to_lds(g0, g1, z4, z4, 0);
#endif
        __builtin_amdgcn_s_wait_tensorcnt((short)0);
    }
    __syncthreads();
#else
    for (int t = threadIdx.x; t < 32 * KD; t += 256) {
        int r = t / KD, k = t - r * KD;
        int gr = rowBase + r; if (gr >= nrows) gr = nrows - 1;
        atile[r * LDA + k] = A[(long)gr * KD + k];
    }
    __syncthreads();
#endif

    const float* ar0 = atile + m * LDA;          // tile 0: rows rowBase..rowBase+15
    const float* ar1 = atile + (16 + m) * LDA;   // tile 1: rows rowBase+16..rowBase+31
    const __bf16* bp = pkW + ((long)wave * 32 + lane) * 16;
    v8f acc0 = {}, acc1 = {};

    for (int kb = 0; kb < KD; kb += 32) {
        const int ka = kb + khalf * 8;

        float4 p0 = *(const float4*)(ar0 + ka);
        float4 p1 = *(const float4*)(ar0 + ka + 4);
        float4 p2 = *(const float4*)(ar0 + ka + 16);
        float4 p3 = *(const float4*)(ar0 + ka + 20);
        float4 q0 = *(const float4*)(ar1 + ka);
        float4 q1 = *(const float4*)(ar1 + ka + 4);
        float4 q2 = *(const float4*)(ar1 + ka + 16);
        float4 q3 = *(const float4*)(ar1 + ka + 20);
        if (RELU) {
            p0.x = fmaxf(p0.x, 0.f); p0.y = fmaxf(p0.y, 0.f); p0.z = fmaxf(p0.z, 0.f); p0.w = fmaxf(p0.w, 0.f);
            p1.x = fmaxf(p1.x, 0.f); p1.y = fmaxf(p1.y, 0.f); p1.z = fmaxf(p1.z, 0.f); p1.w = fmaxf(p1.w, 0.f);
            p2.x = fmaxf(p2.x, 0.f); p2.y = fmaxf(p2.y, 0.f); p2.z = fmaxf(p2.z, 0.f); p2.w = fmaxf(p2.w, 0.f);
            p3.x = fmaxf(p3.x, 0.f); p3.y = fmaxf(p3.y, 0.f); p3.z = fmaxf(p3.z, 0.f); p3.w = fmaxf(p3.w, 0.f);
            q0.x = fmaxf(q0.x, 0.f); q0.y = fmaxf(q0.y, 0.f); q0.z = fmaxf(q0.z, 0.f); q0.w = fmaxf(q0.w, 0.f);
            q1.x = fmaxf(q1.x, 0.f); q1.y = fmaxf(q1.y, 0.f); q1.z = fmaxf(q1.z, 0.f); q1.w = fmaxf(q1.w, 0.f);
            q2.x = fmaxf(q2.x, 0.f); q2.y = fmaxf(q2.y, 0.f); q2.z = fmaxf(q2.z, 0.f); q2.w = fmaxf(q2.w, 0.f);
            q3.x = fmaxf(q3.x, 0.f); q3.y = fmaxf(q3.y, 0.f); q3.z = fmaxf(q3.z, 0.f); q3.w = fmaxf(q3.w, 0.f);
        }

        v16bf a0, a1;
        a0[0]  = (__bf16)p0.x; a0[1]  = (__bf16)p0.y; a0[2]  = (__bf16)p0.z; a0[3]  = (__bf16)p0.w;
        a0[4]  = (__bf16)p1.x; a0[5]  = (__bf16)p1.y; a0[6]  = (__bf16)p1.z; a0[7]  = (__bf16)p1.w;
        a0[8]  = (__bf16)p2.x; a0[9]  = (__bf16)p2.y; a0[10] = (__bf16)p2.z; a0[11] = (__bf16)p2.w;
        a0[12] = (__bf16)p3.x; a0[13] = (__bf16)p3.y; a0[14] = (__bf16)p3.z; a0[15] = (__bf16)p3.w;
        a1[0]  = (__bf16)q0.x; a1[1]  = (__bf16)q0.y; a1[2]  = (__bf16)q0.z; a1[3]  = (__bf16)q0.w;
        a1[4]  = (__bf16)q1.x; a1[5]  = (__bf16)q1.y; a1[6]  = (__bf16)q1.z; a1[7]  = (__bf16)q1.w;
        a1[8]  = (__bf16)q2.x; a1[9]  = (__bf16)q2.y; a1[10] = (__bf16)q2.z; a1[11] = (__bf16)q2.w;
        a1[12] = (__bf16)q3.x; a1[13] = (__bf16)q3.y; a1[14] = (__bf16)q3.z; a1[15] = (__bf16)q3.w;

        v16bf b = *(const v16bf*)bp;          // 32 contiguous bytes, shared by both tiles
        bp += 8 * 32 * 16;                    // next k-block

        acc0 = __builtin_amdgcn_wmma_f32_16x16x32_bf16(false, a0, false, b, (short)0, acc0, false, false);
        acc1 = __builtin_amdgcn_wmma_f32_16x16x32_bf16(false, a1, false, b, (short)0, acc1, false, false);
    }

    const int col = wave * 16 + m;
    if (rowBase + 32 <= nrows) {
#pragma unroll
        for (int v = 0; v < 8; ++v)
            out[(long)(rowBase + v + 8 * khalf) * DH + col] = acc0[v];
#pragma unroll
        for (int v = 0; v < 8; ++v)
            out[(long)(rowBase + 16 + v + 8 * khalf) * DH + col] = acc1[v];
    } else {
#pragma unroll
        for (int v = 0; v < 8; ++v) {
            int r0 = rowBase + v + 8 * khalf;
            if (r0 < nrows) out[(long)r0 * DH + col] = acc0[v];
            int r1 = rowBase + 16 + v + 8 * khalf;
            if (r1 < nrows) out[(long)r1 * DH + col] = acc1[v];
        }
    }
}

// ---------------- self-loop + bias init: agg[i,:] = lin[i,:]*dinv[i]^2 + b ----------------

__global__ void init_agg_kernel(const float* __restrict__ lin, const float* __restrict__ dinv,
                                const float* __restrict__ bias, float* __restrict__ agg, int n) {
    long t = (long)blockIdx.x * blockDim.x + threadIdx.x;
    int node = (int)(t >> 5);
    int c4   = (int)(t & 31) * 4;
    if (node >= n) return;
    float s = dinv[node]; s = s * s;
    const float4 l  = *(const float4*)(lin  + (long)node * DH + c4);
    const float4 bb = *(const float4*)(bias + c4);
    float4 o;
    o.x = l.x * s + bb.x; o.y = l.y * s + bb.y;
    o.z = l.z * s + bb.z; o.w = l.w * s + bb.w;
    *(float4*)(agg + (long)node * DH + c4) = o;
}

// ---------------- edge scatter: agg[dst,:] += lin[src,:] * dinv[src]*dinv[dst] ----------------

__global__ void scatter_edges_kernel(const int* __restrict__ src, const int* __restrict__ dst,
                                     const float* __restrict__ lin, const float* __restrict__ dinv,
                                     float* __restrict__ agg, int E) {
    long t = (long)blockIdx.x * blockDim.x + threadIdx.x;
    int e  = (int)(t >> 5);
    int c4 = (int)(t & 31) * 4;
    if (e >= E) return;
    int s = src[e], d = dst[e];
    float w = dinv[s] * dinv[d];
    const float4 l = *(const float4*)(lin + (long)s * DH + c4);
    float* ap = agg + (long)d * DH + c4;
    unsafeAtomicAdd(ap + 0, l.x * w);
    unsafeAtomicAdd(ap + 1, l.y * w);
    unsafeAtomicAdd(ap + 2, l.z * w);
    unsafeAtomicAdd(ap + 3, l.w * w);
}

// ---------------- decode: out[e] = dot(z[s], z[d]) over 128 dims ----------------

__global__ void decode_dot_kernel(const int* __restrict__ sidx, const int* __restrict__ didx,
                                  const float* __restrict__ z, float* __restrict__ out, int EL) {
    long t = (long)blockIdx.x * blockDim.x + threadIdx.x;
    int e    = (int)(t >> 5);
    int lane = (int)(t & 31);
    if (e >= EL) return;
    int s = sidx[e], d = didx[e];
    const float4 a = *(const float4*)(z + (long)s * DH + lane * 4);
    const float4 b = *(const float4*)(z + (long)d * DH + lane * 4);
    float v = a.x * b.x + a.y * b.y + a.z * b.z + a.w * b.w;
#pragma unroll
    for (int off = 16; off > 0; off >>= 1) v += __shfl_xor(v, off, 32);
    if (lane == 0) out[e] = v;
}

// ---------------- host launch ----------------

extern "C" void kernel_launch(void* const* d_in, const int* in_sizes, int n_in,
                              void* d_out, int out_size, void* d_ws, size_t ws_size,
                              hipStream_t stream) {
    const float* x   = (const float*)d_in[0];
    const int*   ei  = (const int*)d_in[1];
    const int*   eli = (const int*)d_in[2];
    const float* W1  = (const float*)d_in[3];
    const float* b1  = (const float*)d_in[4];
    const float* W2  = (const float*)d_in[5];
    const float* b2  = (const float*)d_in[6];
    float* out = (float*)d_out;

    const int N  = in_sizes[0] / DIN;
    const int E  = in_sizes[1] / 2;
    const int EL = in_sizes[2] / 2;
    const int* src = ei;
    const int* dst = ei + E;
    const int* ls  = eli;
    const int* ld  = eli + EL;

    // workspace: dinv [N] | lin [N*DH] | agg [N*DH] | pkW1 | pkW2   (~103 MB)
    char* ws = (char*)d_ws;
    size_t off = 0;
    float*  dinv = (float*)(ws + off);  off += ((size_t)N * 4 + 255) & ~(size_t)255;
    float*  lin  = (float*)(ws + off);  off += ((size_t)N * DH * 4 + 255) & ~(size_t)255;
    float*  agg  = (float*)(ws + off);  off += ((size_t)N * DH * 4 + 255) & ~(size_t)255;
    __bf16* pk1  = (__bf16*)(ws + off); off += ((size_t)DIN * DH * 2 + 255) & ~(size_t)255;
    __bf16* pk2  = (__bf16*)(ws + off);

    const int TPB = 256;
    dim3 blk(TPB);

    // degree -> dinv
    zero_f32_kernel<<<dim3((N + TPB - 1) / TPB), blk, 0, stream>>>(dinv, N);
    count_deg_kernel<<<dim3((E + TPB - 1) / TPB), blk, 0, stream>>>(dst, dinv, E);
    deg_to_dinv_kernel<<<dim3((N + TPB - 1) / TPB), blk, 0, stream>>>(dinv, N);

    // pre-pack weights (bf16, fragment order)
    pack_w_kernel<<<dim3((DIN * DH + TPB - 1) / TPB), blk, 0, stream>>>(W1, pk1, DIN);
    pack_w_kernel<<<dim3((DH * DH + TPB - 1) / TPB), blk, 0, stream>>>(W2, pk2, DH);

    const unsigned gemmGrid = (unsigned)((N + 31) / 32);
    const unsigned nodeGrid = (unsigned)(((long)N * 32 + TPB - 1) / TPB);
    const unsigned edgeGrid = (unsigned)(((long)E * 32 + TPB - 1) / TPB);
    const unsigned declGrid = (unsigned)(((long)EL * 32 + TPB - 1) / TPB);

    // layer 1
    gemm_wmma_bf16_kernel<DIN, false><<<dim3(gemmGrid), blk, 0, stream>>>(x, pk1, lin, N);
    init_agg_kernel<<<dim3(nodeGrid), blk, 0, stream>>>(lin, dinv, b1, agg, N);
    scatter_edges_kernel<<<dim3(edgeGrid), blk, 0, stream>>>(src, dst, lin, dinv, agg, E);

    // layer 2 (relu fused into A-load)
    gemm_wmma_bf16_kernel<DH, true><<<dim3(gemmGrid), blk, 0, stream>>>(agg, pk2, lin, N);
    init_agg_kernel<<<dim3(nodeGrid), blk, 0, stream>>>(lin, dinv, b2, agg, N);
    scatter_edges_kernel<<<dim3(edgeGrid), blk, 0, stream>>>(src, dst, lin, dinv, agg, E);

    // decode
    decode_dot_kernel<<<dim3(declGrid), blk, 0, stream>>>(ls, ld, agg, out, EL);
}
